// HeadPac_67018669687229
// MI455X (gfx1250) — matrix operand order repaired and checked by reading, not compiled, not run on hardware
//
#include <hip/hip_runtime.h>

typedef __attribute__((ext_vector_type(16))) _Float16 v16h;
typedef __attribute__((ext_vector_type(8)))  float    v8f;

// ---------------------------------------------------------------------------
// Weight repack: fp32 OIHW -> fp16 in per-lane WMMA A-fragment order.
// K index is tap-major: k = t*C + c  (t = dy*3+dx), zero-padded to mult of 32.
// Layout: wpk[mt][kt][lane(32)][16 halves]; lane L holds M = L&15 with the
// contiguous K order (same as the B tile): lanes<16 -> K in {0..15},
// lanes>=16 -> K in {16..31} (slot s holds K = 16*half + s).
// ---------------------------------------------------------------------------
__global__ void repack_w_kernel(const float* __restrict__ w,
                                _Float16* __restrict__ wpk,
                                int C, int O, int ktiles) {
  int idx = blockIdx.x * blockDim.x + threadIdx.x;
  int total = (O / 16) * ktiles * 32 * 16;
  if (idx >= total) return;
  int s    = idx & 15;          // half slot within lane (0..15)
  int lane = (idx >> 4) & 31;
  int kt   = (idx >> 9) % ktiles;
  int mt   = (idx >> 9) / ktiles;

  int v = s >> 1, p = s & 1;
  int half_sel = lane >> 4;
  int klocal = half_sel * 16 + s;
  int k  = kt * 32 + klocal;
  int oc = mt * 16 + (lane & 15);

  int K9 = 9 * C;
  float val = 0.f;
  if (k < K9) {
    int t = k / C;
    int c = k - t * C;
    int dy = t / 3, dx = t - dy * 3;
    val = w[((size_t)(oc * C + c) * 3 + dy) * 3 + dx];
  }
  wpk[idx] = (_Float16)val;
}

// ---------------------------------------------------------------------------
// Implicit-GEMM 3x3 conv (pad=1, stride 1 or 2), optional PAC per-pixel tap
// scale folded into the B tile, fused bias + ReLU.
//   Per K-chunk:  D[16 oc x 16 px] += A[16 oc x 32 k] * B[32 k x 16 px]
// Block = 128 threads = 4 waves; one cooperative 32x16 B tile (double
// buffered LDS) consumed by 4 waves x MW M-tiles = 4*MW WMMAs per chunk.
// UNIF: C is a multiple of 32, so each 32-K chunk sits inside one tap ->
// tap id, coords, bounds and PAC scale are chunk-uniform (no per-element
// decode, no integer division anywhere in the hot loop).
// shTX = log2(oW/16), shOH = log2(oH), shC = log2(C) (all powers of two).
// ---------------------------------------------------------------------------
template <int MW, bool UNIF>
__global__ __launch_bounds__(128) void conv_wmma_kernel(
    const float* __restrict__ x, const _Float16* __restrict__ wpk,
    const float* __restrict__ bias, const float* __restrict__ kern,
    float* __restrict__ out,
    int C, int shC, int H, int W, int O, int oH, int oW,
    int shTX, int shOH, int stride, int ktiles)
{
  __shared__ __align__(64) _Float16 Bt[2][16][32];   // [buf][n][k] 2 KB

  const int tid  = threadIdx.x;
  const int lane = tid & 31;
  const int wave = tid >> 5;

  // Block -> (b, oy, ox0) via shifts only
  const int px  = blockIdx.x;
  const int ox0 = (px & ((1 << shTX) - 1)) << 4;
  const int tmp = px >> shTX;
  const int oy  = tmp & (oH - 1);
  const int b   = tmp >> shOH;
  const int mt0 = (blockIdx.y * 4 + wave) * MW;

  const int    K9 = 9 * C;
  const size_t HW = (size_t)H * W;

  // B-tile build role: this thread builds column n_b, rows kk0 + 8*i.
  const int n_b  = tid & 15;
  const int kk0  = tid >> 4;                 // 0..7
  const int ox_b = ox0 + n_b;
  const int iyc  = oy * stride - 1;
  const int ixc  = ox_b * stride - 1;
  const float* kernp = kern
      ? (kern + ((size_t)b * 9) * HW + (size_t)(oy * stride) * W + (size_t)(ox_b * stride))
      : nullptr;

  // WMMA fragment lane decode
  const int nl = lane & 15;                  // B/D column
  const int hs = lane >> 4;                  // K-half / M+8 selector

  v8f acc[MW];
  #pragma unroll
  for (int m = 0; m < MW; ++m) acc[m] = (v8f){};

  for (int kt = 0; kt < ktiles; ++kt) {
    const int buf = kt & 1;
    if (UNIF) {
      // Chunk-uniform tap decode
      const int  k0 = kt << 5;
      const int  t  = k0 >> shC;
      const int  dy = (t * 11) >> 5;         // t/3 for t in [0,9)
      const int  dx = t - dy * 3;
      const int  iy = iyc + dy, ix = ixc + dx;
      const bool valid = (unsigned)iy < (unsigned)H && (unsigned)ix < (unsigned)W;
      const float ks = kernp ? kernp[(size_t)t * HW] : 1.f;
      const int  c0 = k0 & (C - 1);
      const float* p = x + (((size_t)b * C + c0) * H + iy) * W + ix;
      #pragma unroll
      for (int i = 0; i < 4; ++i) {
        int kk = kk0 + i * 8;
        float val = valid ? p[(size_t)kk * HW] : 0.f;
        Bt[buf][n_b][kk] = (_Float16)(val * ks);
      }
    } else {
      // Generic (only used for C==1 front layer): per-element decode, shifts.
      #pragma unroll
      for (int i = 0; i < 4; ++i) {
        int kk = kk0 + i * 8;
        int k  = (kt << 5) + kk;
        float val = 0.f;
        if (k < K9) {
          int t  = k >> shC;
          int c  = k & (C - 1);
          int dy = (t * 11) >> 5;
          int dx = t - dy * 3;
          int iy = iyc + dy, ix = ixc + dx;
          if ((unsigned)iy < (unsigned)H && (unsigned)ix < (unsigned)W)
            val = x[(((size_t)b * C + c) * H + iy) * W + ix];
          if (kernp) val *= kernp[(size_t)t * HW];
        }
        Bt[buf][n_b][kk] = (_Float16)val;
      }
    }
    __syncthreads();

    // B fragment: lane = column nl, 16 contiguous K halves (32B aligned)
    v16h bm = *(const v16h*)(&Bt[buf][nl][hs * 16]);
    #pragma unroll
    for (int m = 0; m < MW; ++m) {
      v16h a = *(const v16h*)(wpk + ((((size_t)(mt0 + m) * ktiles + kt) * 32) + lane) * 16);
      acc[m] = __builtin_amdgcn_wmma_f32_16x16x32_f16(
          /*neg_a=*/false, a, /*neg_b=*/false, bm,
          /*c_mod=*/(short)0, acc[m], /*reuse_a=*/false, /*reuse_b=*/false);
    }
  }

  // D layout: lane col = nl; VGPR v holds row M = v (lanes<16) / v+8 (lanes>=16)
  const int ox = ox0 + nl;
  const size_t plane = (size_t)oH * oW;
  const size_t obase = (((size_t)b * O) * oH + oy) * oW + ox;
  #pragma unroll
  for (int m = 0; m < MW; ++m) {
    #pragma unroll
    for (int v = 0; v < 8; ++v) {
      int oc = (mt0 + m) * 16 + v + hs * 8;
      float r = acc[m][v] + bias[oc];
      out[obase + (size_t)oc * plane] = r > 0.f ? r : 0.f;
    }
  }
}

// ---------------------------------------------------------------------------
// PAC Gaussian kernel: k[b,t,y,x] = exp(-0.5 * 1e-8 * sum_c (h[nbr]-h[ctr])^2)
// (GS = 1e-4 folded in as 1e-8 on the squared distance). Zero padding.
// ---------------------------------------------------------------------------
__global__ void pac_kernel_kernel(const float* __restrict__ h,
                                  float* __restrict__ kern,
                                  int Bn, int C, int H, int W) {
  int idx = blockIdx.x * blockDim.x + threadIdx.x;
  int total = Bn * H * W;
  if (idx >= total) return;
  int x  = idx & (W - 1);           // W is a power of two
  int t2 = idx / W;
  int y  = t2 & (H - 1);            // H is a power of two
  int b  = t2 / H;

  float s[9];
  #pragma unroll
  for (int t = 0; t < 9; ++t) s[t] = 0.f;

  for (int c = 0; c < C; ++c) {
    const float* plane = h + (((size_t)b * C + c) * H) * W;
    float g = plane[(size_t)y * W + x];
    #pragma unroll
    for (int t = 0; t < 9; ++t) {
      int dy = t / 3 - 1, dx = t % 3 - 1;
      int yy = y + dy, xx = x + dx;
      float v = 0.f;
      if ((unsigned)yy < (unsigned)H && (unsigned)xx < (unsigned)W)
        v = plane[(size_t)yy * W + xx];
      float d = v - g;
      s[t] += d * d;
    }
  }
  #pragma unroll
  for (int t = 0; t < 9; ++t)
    kern[(((size_t)b * 9 + t) * H + y) * W + x] = __expf(-0.5e-8f * s[t]);
}

static inline int ilog2i(int v) { int s = 0; while ((1 << s) < v) ++s; return s; }

// ---------------------------------------------------------------------------
// Host-side orchestration
// ---------------------------------------------------------------------------
extern "C" void kernel_launch(void* const* d_in, const int* in_sizes, int n_in,
                              void* d_out, int out_size, void* d_ws, size_t ws_size,
                              hipStream_t stream) {
  (void)in_sizes; (void)n_in; (void)out_size; (void)ws_size;
  const int Bn = 8;
  const float* x_in = (const float*)d_in[0];

  // Workspace layout (bytes)
  char* ws = (char*)d_ws;
  float*    actA = (float*)(ws + 0);                       // 134,217,728 B
  float*    actB = (float*)(ws + 134217728);               // 134,217,728 B
  float*    kbuf = (float*)(ws + 268435456);               //   4,718,592 B
  _Float16* wpk  = (_Float16*)(ws + 273154048);            //  18,874,368 B

  auto conv_layer = [&](const float* in, int li, int C, int O, int H, int W,
                        int stride, const float* kern, float* out) {
    const float* w  = (const float*)d_in[1 + 2 * li];
    const float* bs = (const float*)d_in[2 + 2 * li];
    int K9 = 9 * C;
    int ktiles = (K9 + 31) / 32;
    int totalpk = (O / 16) * ktiles * 512;
    repack_w_kernel<<<(totalpk + 255) / 256, 256, 0, stream>>>(w, wpk, C, O, ktiles);

    int oH = (H - 1) / stride + 1;
    int oW = (W - 1) / stride + 1;
    int shTX = ilog2i(oW / 16);
    int shOH = ilog2i(oH);
    int shC  = ilog2i(C);
    if (C >= 32) {
      if (O >= 128) {
        dim3 grid((unsigned)(Bn * oH * (oW / 16)), (unsigned)(O / 128));
        conv_wmma_kernel<2, true><<<grid, 128, 0, stream>>>(
            in, wpk, bs, kern, out, C, shC, H, W, O, oH, oW, shTX, shOH, stride, ktiles);
      } else {
        dim3 grid((unsigned)(Bn * oH * (oW / 16)), (unsigned)(O / 64));
        conv_wmma_kernel<1, true><<<grid, 128, 0, stream>>>(
            in, wpk, bs, kern, out, C, shC, H, W, O, oH, oW, shTX, shOH, stride, ktiles);
      }
    } else {
      dim3 grid((unsigned)(Bn * oH * (oW / 16)), (unsigned)(O / 64));
      conv_wmma_kernel<1, false><<<grid, 128, 0, stream>>>(
          in, wpk, bs, kern, out, C, shC, H, W, O, oH, oW, shTX, shOH, stride, ktiles);
    }
  };
  auto pac = [&](const float* hsrc, int C, int H, int W) {
    int total = Bn * H * W;
    pac_kernel_kernel<<<(total + 255) / 256, 256, 0, stream>>>(hsrc, kbuf, Bn, C, H, W);
  };

  // Layer index li matches params order: c1_1,c1_2,c1_3,p2_1..p5_2
  // --- level 1 (plain convs, 256x256) ---
  conv_layer(x_in, 0,   1,   64, 256, 256, 1, nullptr, actA);   // c1_1
  conv_layer(actA, 1,  64,   64, 256, 256, 1, nullptr, actB);   // c1_2
  conv_layer(actB, 2,  64,   64, 256, 256, 2, nullptr, actA);   // c1_3 -> h2 128x128

  // --- level 2 (PAC, 128x128) ---
  pac(actA, 64, 128, 128);                                       // k2
  conv_layer(actA, 3,  64,  128, 128, 128, 1, kbuf, actB);       // p2_1
  conv_layer(actB, 4, 128,  128, 128, 128, 1, kbuf, actA);       // p2_2
  conv_layer(actA, 5, 128,  128, 128, 128, 2, kbuf, actB);       // p2_3 -> h3 64x64

  // --- level 3 (PAC, 64x64) ---
  pac(actB, 128, 64, 64);                                        // k3
  conv_layer(actB, 6, 128,  256,  64,  64, 1, kbuf, actA);       // p3_1
  conv_layer(actA, 7, 256,  256,  64,  64, 1, kbuf, actB);       // p3_2
  conv_layer(actB, 8, 256,  256,  64,  64, 2, kbuf, actA);       // p3_3 -> h4 32x32

  // --- level 4 (PAC, 32x32) ---
  pac(actA, 256, 32, 32);                                        // k4
  conv_layer(actA,  9, 256, 512,  32,  32, 1, kbuf, actB);       // p4_1
  conv_layer(actB, 10, 512, 512,  32,  32, 1, kbuf, actA);       // p4_2
  conv_layer(actA, 11, 512, 512,  32,  32, 2, kbuf, actB);       // p4_3 -> h5 16x16

  // --- level 5 (PAC, 16x16) ---
  pac(actB, 512, 16, 16);                                        // k5
  conv_layer(actB, 12,  512, 1024, 16, 16, 1, kbuf, actA);       // p5_1
  conv_layer(actA, 13, 1024, 1024, 16, 16, 1, kbuf, (float*)d_out); // p5_2
}
